// LongcatFlashMoE_2731599200768
// MI455X (gfx1250) — compile-verified, matrix operand order checked by reading, not run on hardware
//
#include <hip/hip_runtime.h>
#include <math.h>

#define T_TOK   2048
#define HIDDEN  2048
#define FFN     1024
#define TWO_FFN 2048
#define NE      16
#define NTOT    32
#define TOPK    4
#define RSCALE  2.5f

// fragment-set geometry: one WMMA operand set = 32 lanes x 16 bf16 = 512 elems (1KB)
#define FRAG_ELEMS 512
#define HB32 (HIDDEN / 32)   // 64  K-blocks over HIDDEN
#define FB32 (FFN / 32)      // 32  K-blocks over FFN
#define NB13 (TWO_FFN / 16)  // 128 N-blocks of w13 (gate: 0..63, up: 64..127)
#define NB2  (HIDDEN / 16)   // 128 N-blocks of w2
#define TB   (T_TOK / 16)    // 128 token tiles

typedef __attribute__((ext_vector_type(16))) __bf16 v16bf;
typedef __attribute__((ext_vector_type(8)))  float  v8f;

__device__ __forceinline__ unsigned short f32_to_bf16(float f) {
    union { float f; unsigned int u; } v; v.f = f;
    unsigned int r = v.u + 0x7FFFu + ((v.u >> 16) & 1u);   // RNE
    return (unsigned short)(r >> 16);
}

// ---------------------------------------------------------------------------
// Router: one wave32 per token; lane == expert (NTOT == 32 == wave width).
// ---------------------------------------------------------------------------
__global__ void router_kernel(const float* __restrict__ hs,
                              const float* __restrict__ rw,
                              const float* __restrict__ bias,
                              float* __restrict__ comb,
                              float* __restrict__ ztot) {
    const int t = blockIdx.x, lane = threadIdx.x;
    const float4* hv = (const float4*)(hs + (size_t)t * HIDDEN);
    const float4* wv = (const float4*)(rw + (size_t)lane * HIDDEN);
    float acc = 0.f;
    #pragma unroll 4
    for (int i = 0; i < HIDDEN / 4; ++i) {
        float4 a = hv[i], b = wv[i];
        acc += a.x * b.x + a.y * b.y + a.z * b.z + a.w * b.w;
    }
    float m = acc;
    #pragma unroll
    for (int off = 16; off > 0; off >>= 1) m = fmaxf(m, __shfl_xor(m, off, 32));
    float e = __expf(acc - m);
    float s = e;
    #pragma unroll
    for (int off = 16; off > 0; off >>= 1) s += __shfl_xor(s, off, 32);
    float score = e / s + bias[lane];

    float mycomb = 0.f, zt = 0.f;
    #pragma unroll
    for (int k = 0; k < TOPK; ++k) {
        float v = score; int idx = lane;
        #pragma unroll
        for (int off = 16; off > 0; off >>= 1) {
            float ov = __shfl_xor(v, off, 32);
            int   oi = __shfl_xor(idx, off, 32);
            if (ov > v || (ov == v && oi < idx)) { v = ov; idx = oi; }
        }
        float wsel = v * RSCALE;
        if (idx >= NE) zt += wsel;
        else if (lane == idx) mycomb += wsel;
        if (lane == idx) score = -1e30f;
    }
    if (lane < NE) comb[(size_t)t * NE + lane] = mycomb;
    if (lane == 0) ztot[t] = zt;
}

// ---------------------------------------------------------------------------
// One-shot cvt+swizzle: fp32 -> bf16 in this file's WMMA fragment order (contiguous K per lane half).
// A-frag (16x32): lane l -> row m=l&15, K-half (l>>4)*16, 16 contig K.
// B-frag (32x16): lane l -> col n=l&15, K-half (l>>4)*16, 16 contig K.
// Fragment set stored [set][lane][16] => 1KB contiguous per wave-load.
// ---------------------------------------------------------------------------
__global__ void convert_hidden(const float* __restrict__ hs,
                               unsigned short* __restrict__ hA) {
    int set = blockIdx.x * 8 + (threadIdx.x >> 5);
    int lane = threadIdx.x & 31;
    int tblk = set / HB32, kblk = set % HB32;
    int m = lane & 15, koff = (lane >> 4) << 4;
    const float* src = hs + (size_t)(tblk * 16 + m) * HIDDEN + kblk * 32 + koff;
    __align__(16) unsigned short tmp[16];
    #pragma unroll
    for (int j = 0; j < 16; j += 4) {
        float4 v = *(const float4*)(src + j);
        tmp[j+0] = f32_to_bf16(v.x); tmp[j+1] = f32_to_bf16(v.y);
        tmp[j+2] = f32_to_bf16(v.z); tmp[j+3] = f32_to_bf16(v.w);
    }
    unsigned short* dst = hA + (size_t)set * FRAG_ELEMS + lane * 16;
    *(uint4*)(dst)     = *(const uint4*)(tmp);
    *(uint4*)(dst + 8) = *(const uint4*)(tmp + 8);
}

__global__ void convert_w13(const float* __restrict__ w13,
                            unsigned short* __restrict__ wB) {
    int set = blockIdx.x * 8 + (threadIdx.x >> 5);
    int lane = threadIdx.x & 31;
    int e = set / (NB13 * HB32);
    int rem = set % (NB13 * HB32);
    int fblk = rem / HB32, kblk = rem % HB32;
    int n = lane & 15, koff = (lane >> 4) << 4;
    const float* src = w13 + ((size_t)e * HIDDEN + kblk * 32 + koff) * TWO_FFN
                           + fblk * 16 + n;
    __align__(16) unsigned short tmp[16];
    #pragma unroll
    for (int j = 0; j < 16; ++j) tmp[j] = f32_to_bf16(src[(size_t)j * TWO_FFN]);
    unsigned short* dst = wB + (size_t)set * FRAG_ELEMS + lane * 16;
    *(uint4*)(dst)     = *(const uint4*)(tmp);
    *(uint4*)(dst + 8) = *(const uint4*)(tmp + 8);
}

__global__ void convert_w2(const float* __restrict__ w2,
                           unsigned short* __restrict__ wB) {
    int set = blockIdx.x * 8 + (threadIdx.x >> 5);
    int lane = threadIdx.x & 31;
    int e = set / (NB2 * FB32);
    int rem = set % (NB2 * FB32);
    int hblk = rem / FB32, kblk = rem % FB32;
    int n = lane & 15, koff = (lane >> 4) << 4;
    const float* src = w2 + ((size_t)e * FFN + kblk * 32 + koff) * HIDDEN
                          + hblk * 16 + n;
    __align__(16) unsigned short tmp[16];
    #pragma unroll
    for (int j = 0; j < 16; ++j) tmp[j] = f32_to_bf16(src[(size_t)j * HIDDEN]);
    unsigned short* dst = wB + (size_t)set * FRAG_ELEMS + lane * 16;
    *(uint4*)(dst)     = *(const uint4*)(tmp);
    *(uint4*)(dst + 8) = *(const uint4*)(tmp + 8);
}

// ---------------------------------------------------------------------------
// Stage 1: act[e,t,f] = comb[t,e] * silu(gate) * up,  bf16, fragment layout.
// Register-blocked 2(M-tiles) x (2 gate + 2 up) per wave, 8 WMMA / k-step.
// grid (FFN/16/2/8 = 4, TB/2 = 64, NE), block 256 (8 waves).
// Pre-scaling by comb makes unused-token rows exactly zero; fully-unused
// (expert, 32-token) blocks zero-fill their act slice and exit.
// ---------------------------------------------------------------------------
__global__ void moe_stage1(const unsigned short* __restrict__ hA,
                           const unsigned short* __restrict__ wB13,
                           const float* __restrict__ comb,
                           unsigned short* __restrict__ act) {
    const int e = blockIdx.z;
    const int tblk0 = blockIdx.y * 2;
    const int tid = threadIdx.x, lane = tid & 31, wv = tid >> 5;
    const int jg = (blockIdx.x * 8 + wv) * 2;        // gate N-blocks jg, jg+1

    bool u0 = false, u1 = false;
    #pragma unroll
    for (int j = 0; j < 16; ++j) {
        u0 |= (comb[(size_t)(tblk0 * 16 + j) * NE + e] != 0.f);
        u1 |= (comb[(size_t)(tblk0 * 16 + 16 + j) * NE + e] != 0.f);
    }
    if (!u0 && !u1) {                                 // zero-fill + exit
        const uint4 z = {0u, 0u, 0u, 0u};
        #pragma unroll
        for (int mt = 0; mt < 2; ++mt) {
            uint4* dst = (uint4*)(act + (((size_t)e * TB + tblk0 + mt) * FB32
                                         + blockIdx.x * 8) * FRAG_ELEMS);
            for (int c = tid; c < 512; c += 256) dst[c] = z;
        }
        return;
    }

    const size_t lo = (size_t)lane * 16;
    size_t aB[2], gB[2], uB[2];
    #pragma unroll
    for (int mt = 0; mt < 2; ++mt)
        aB[mt] = ((size_t)(tblk0 + mt) * HB32) * FRAG_ELEMS + lo;
    #pragma unroll
    for (int q = 0; q < 2; ++q) {
        gB[q] = (((size_t)e * NB13 + jg + q)      * HB32) * FRAG_ELEMS + lo;
        uB[q] = (((size_t)e * NB13 + jg + q + 64) * HB32) * FRAG_ELEMS + lo;
    }

    v8f accg[2][2] = {}, accu[2][2] = {};
    #pragma unroll 2
    for (int k0 = 0; k0 < HB32; ++k0) {
        const size_t ko = (size_t)k0 * FRAG_ELEMS;
        v16bf a0 = *(const v16bf*)(hA + aB[0] + ko);
        v16bf a1 = *(const v16bf*)(hA + aB[1] + ko);
        v16bf bg0 = *(const v16bf*)(wB13 + gB[0] + ko);
        v16bf bg1 = *(const v16bf*)(wB13 + gB[1] + ko);
        v16bf bu0 = *(const v16bf*)(wB13 + uB[0] + ko);
        v16bf bu1 = *(const v16bf*)(wB13 + uB[1] + ko);
        accg[0][0] = __builtin_amdgcn_wmma_f32_16x16x32_bf16(false, a0, false, bg0, (short)0, accg[0][0], false, false);
        accg[0][1] = __builtin_amdgcn_wmma_f32_16x16x32_bf16(false, a0, false, bg1, (short)0, accg[0][1], false, false);
        accu[0][0] = __builtin_amdgcn_wmma_f32_16x16x32_bf16(false, a0, false, bu0, (short)0, accu[0][0], false, false);
        accu[0][1] = __builtin_amdgcn_wmma_f32_16x16x32_bf16(false, a0, false, bu1, (short)0, accu[0][1], false, false);
        accg[1][0] = __builtin_amdgcn_wmma_f32_16x16x32_bf16(false, a1, false, bg0, (short)0, accg[1][0], false, false);
        accg[1][1] = __builtin_amdgcn_wmma_f32_16x16x32_bf16(false, a1, false, bg1, (short)0, accg[1][1], false, false);
        accu[1][0] = __builtin_amdgcn_wmma_f32_16x16x32_bf16(false, a1, false, bu0, (short)0, accu[1][0], false, false);
        accu[1][1] = __builtin_amdgcn_wmma_f32_16x16x32_bf16(false, a1, false, bu1, (short)0, accu[1][1], false, false);
    }

    // swiglu * comb -> bf16 act in stage-2 A-fragment layout
    const int hi8 = (lane >> 4) << 3, fl = lane & 15;
    #pragma unroll
    for (int mt = 0; mt < 2; ++mt) {
        const int tbase = (tblk0 + mt) * 16;
        const size_t actB = (((size_t)e * TB + tblk0 + mt) * FB32 + (jg >> 1))
                            * FRAG_ELEMS;
        #pragma unroll
        for (int q = 0; q < 2; ++q) {
            #pragma unroll
            for (int i = 0; i < 8; ++i) {
                int m = i + hi8;
                float cw = comb[(size_t)(tbase + m) * NE + e];
                float g = accg[mt][q][i], u = accu[mt][q][i];
                float aV = (g / (1.f + __expf(-g))) * u * cw;
                act[actB + (size_t)(q * 16 + m) * 16 + fl] = f32_to_bf16(aV);
            }
        }
    }
}

// ---------------------------------------------------------------------------
// Stage 2: out = sum_e (act_e @ w2[e]) + hs*ztot   (act pre-scaled by comb).
// Register-blocked 2(M) x 4(N) per wave, 8 WMMA / k-step, no LDS/barriers.
// grid (NB2/4/8 = 4, TB/2 = 64), block 256 (8 waves).
// ---------------------------------------------------------------------------
__global__ void moe_stage2(const float* __restrict__ hs,
                           const unsigned short* __restrict__ act,
                           const unsigned short* __restrict__ wB2,
                           const float* __restrict__ comb,
                           const float* __restrict__ ztot,
                           float* __restrict__ out) {
    const int tblk0 = blockIdx.y * 2;
    const int tid = threadIdx.x, lane = tid & 31, wv = tid >> 5;
    const int nb0 = (blockIdx.x * 8 + wv) * 4;       // 4 N-blocks of HIDDEN
    const size_t lo = (size_t)lane * 16;
    const int hi8 = (lane >> 4) << 3, fl = lane & 15;

    v8f acc[2][4] = {};
    for (int e = 0; e < NE; ++e) {
        // wave-uniform skip: all 32 token rows unused by expert e
        int r = lane & 15;
        bool any = (comb[(size_t)(tblk0 * 16 + r) * NE + e] != 0.f) ||
                   (comb[(size_t)(tblk0 * 16 + 16 + r) * NE + e] != 0.f);
        if (__ballot(any) == 0ULL) continue;

        size_t aB[2], bB[4];
        #pragma unroll
        for (int mt = 0; mt < 2; ++mt)
            aB[mt] = (((size_t)e * TB + tblk0 + mt) * FB32) * FRAG_ELEMS + lo;
        #pragma unroll
        for (int nj = 0; nj < 4; ++nj)
            bB[nj] = (((size_t)e * NB2 + nb0 + nj) * FB32) * FRAG_ELEMS + lo;

        #pragma unroll 2
        for (int k0 = 0; k0 < FB32; ++k0) {
            const size_t ko = (size_t)k0 * FRAG_ELEMS;
            v16bf a0 = *(const v16bf*)(act + aB[0] + ko);
            v16bf a1 = *(const v16bf*)(act + aB[1] + ko);
            v16bf b0 = *(const v16bf*)(wB2 + bB[0] + ko);
            v16bf b1 = *(const v16bf*)(wB2 + bB[1] + ko);
            v16bf b2 = *(const v16bf*)(wB2 + bB[2] + ko);
            v16bf b3 = *(const v16bf*)(wB2 + bB[3] + ko);
            acc[0][0] = __builtin_amdgcn_wmma_f32_16x16x32_bf16(false, a0, false, b0, (short)0, acc[0][0], false, false);
            acc[0][1] = __builtin_amdgcn_wmma_f32_16x16x32_bf16(false, a0, false, b1, (short)0, acc[0][1], false, false);
            acc[0][2] = __builtin_amdgcn_wmma_f32_16x16x32_bf16(false, a0, false, b2, (short)0, acc[0][2], false, false);
            acc[0][3] = __builtin_amdgcn_wmma_f32_16x16x32_bf16(false, a0, false, b3, (short)0, acc[0][3], false, false);
            acc[1][0] = __builtin_amdgcn_wmma_f32_16x16x32_bf16(false, a1, false, b0, (short)0, acc[1][0], false, false);
            acc[1][1] = __builtin_amdgcn_wmma_f32_16x16x32_bf16(false, a1, false, b1, (short)0, acc[1][1], false, false);
            acc[1][2] = __builtin_amdgcn_wmma_f32_16x16x32_bf16(false, a1, false, b2, (short)0, acc[1][2], false, false);
            acc[1][3] = __builtin_amdgcn_wmma_f32_16x16x32_bf16(false, a1, false, b3, (short)0, acc[1][3], false, false);
        }
    }

    #pragma unroll
    for (int mt = 0; mt < 2; ++mt) {
        #pragma unroll
        for (int i = 0; i < 8; ++i) {
            int t = (tblk0 + mt) * 16 + i + hi8;
            float zw = ztot[t];
            #pragma unroll
            for (int nj = 0; nj < 4; ++nj) {
                int h = (nb0 + nj) * 16 + fl;
                out[(size_t)t * HIDDEN + h] =
                    acc[mt][nj][i] + hs[(size_t)t * HIDDEN + h] * zw;
            }
        }
    }
}

// ---------------------------------------------------------------------------
extern "C" void kernel_launch(void* const* d_in, const int* in_sizes, int n_in,
                              void* d_out, int out_size, void* d_ws, size_t ws_size,
                              hipStream_t stream) {
    const float* hs   = (const float*)d_in[0];   // [T, HIDDEN]
    const float* rw   = (const float*)d_in[1];   // [NTOT, HIDDEN]
    const float* bias = (const float*)d_in[2];   // [NTOT]
    const float* w13  = (const float*)d_in[3];   // [E, HIDDEN, 2*FFN]
    const float* w2   = (const float*)d_in[4];   // [E, FFN, HIDDEN]
    float* out = (float*)d_out;

    // workspace: comb | ztot | hA(bf16) | wB13(bf16) | wB2(bf16) | act(bf16)
    char* ws = (char*)d_ws;
    float* comb = (float*)ws;                    ws += (size_t)T_TOK * NE * 4;
    float* ztot = (float*)ws;                    ws += (size_t)T_TOK * 4;
    unsigned short* hA   = (unsigned short*)ws;  ws += (size_t)T_TOK * HIDDEN * 2;
    unsigned short* wB13 = (unsigned short*)ws;  ws += (size_t)NE * HIDDEN * TWO_FFN * 2;
    unsigned short* wB2  = (unsigned short*)ws;  ws += (size_t)NE * FFN * HIDDEN * 2;
    unsigned short* act  = (unsigned short*)ws;

    router_kernel<<<T_TOK, 32, 0, stream>>>(hs, rw, bias, comb, ztot);

    convert_hidden<<<(TB * HB32) / 8, 256, 0, stream>>>(hs, hA);
    convert_w13<<<(NE * NB13 * HB32) / 8, 256, 0, stream>>>(w13, wB13);
    convert_w2<<<(NE * NB2 * FB32) / 8, 256, 0, stream>>>(w2, wB2);

    dim3 g1(FFN / 16 / 2 / 8, TB / 2, NE);       // (4, 64, 16)
    moe_stage1<<<g1, 256, 0, stream>>>(hA, wB13, comb, act);

    dim3 g2(NB2 / 4 / 8, TB / 2);                // (4, 64)
    moe_stage2<<<g2, 256, 0, stream>>>(hs, act, wB2, comb, ztot, out);
}
